// AttentionBlock_3848290697730
// MI455X (gfx1250) — matrix operand order repaired and checked by reading, not compiled, not run on hardware
//
#include <hip/hip_runtime.h>
#include <hip/hip_bf16.h>

typedef _Float16 v16h __attribute__((ext_vector_type(16)));
typedef _Float16 v8h  __attribute__((ext_vector_type(8)));
typedef float    v8f  __attribute__((ext_vector_type(8)));

#define WMMA_F16(a, b, c) \
  __builtin_amdgcn_wmma_f32_16x16x32_f16(false, (a), false, (b), (short)0, (c), false, false)

// Problem constants
#define NB    4
#define NC    256
#define NPIX  4096        // 64*64
#define NG    8
#define SPLIT 4           // key-dimension splits for flash-decoding
#define KEYS_PER_SPLIT (NPIX / SPLIT)

// ---------------------------------------------------------------------------
// Kernel 1: GroupNorm statistics -> per-(b,c) scale/bias
// ---------------------------------------------------------------------------
__global__ __launch_bounds__(256) void gn_stats_kernel(
    const float* __restrict__ x, const float* __restrict__ gn_w,
    const float* __restrict__ gn_b, float* __restrict__ scale,
    float* __restrict__ bias) {
  const int bg = blockIdx.x;           // 0..31
  const int b = bg >> 3, g = bg & 7;
  const float* p = x + ((size_t)b * NC + (size_t)g * 32) * NPIX;
  const int CNT = 32 * NPIX;
  float s = 0.f, s2 = 0.f;
  for (int i = threadIdx.x; i < CNT; i += 256) {
    float v = p[i];
    s += v; s2 += v * v;
  }
  __shared__ float rs[256], rs2[256];
  rs[threadIdx.x] = s; rs2[threadIdx.x] = s2;
  __syncthreads();
  for (int off = 128; off > 0; off >>= 1) {
    if (threadIdx.x < off) {
      rs[threadIdx.x]  += rs[threadIdx.x + off];
      rs2[threadIdx.x] += rs2[threadIdx.x + off];
    }
    __syncthreads();
  }
  const float mean = rs[0] / (float)CNT;
  const float var  = rs2[0] / (float)CNT - mean * mean;
  const float rstd = rsqrtf(var + 1e-5f);
  if (threadIdx.x < 32) {
    int c = g * 32 + threadIdx.x;
    float sc = rstd * gn_w[c];
    scale[b * NC + c] = sc;
    bias[b * NC + c]  = gn_b[c] - mean * sc;
  }
}

// ---------------------------------------------------------------------------
// Kernel 2: fused GN-apply + QKV GEMM (WMMA).  qkv[m][n] = W[m][:] . h[:][n]
//   q -> qw[b][n][c] (f16, pre-scaled by 1/sqrt(C))
//   k -> kw[b][n][c] (f16)
//   v -> vt[b][c][n] (f16, transposed)
// ---------------------------------------------------------------------------
__global__ __launch_bounds__(128) void qkv_kernel(
    const float* __restrict__ x, const float* __restrict__ qkv_w,
    const float* __restrict__ qkv_b, const float* __restrict__ scale,
    const float* __restrict__ bias, _Float16* __restrict__ qw,
    _Float16* __restrict__ kw, _Float16* __restrict__ vt) {
  const int lane = threadIdx.x & 31;
  const int wid  = threadIdx.x >> 5;
  const int l16  = lane & 15, hf = lane >> 4;
  const int m0   = blockIdx.x * 64 + wid * 16;   // output-channel tile (768 total)
  const int ng0  = blockIdx.y * 64;              // global pixel-column base
  const int b    = ng0 >> 12;
  const int nb0  = ng0 & (NPIX - 1);
  const float* xb  = x + (size_t)b * NC * NPIX;
  const float* scb = scale + b * NC;
  const float* bib = bias + b * NC;

  v8f acc[4] = {};
  for (int kk = 0; kk < NC; kk += 32) {
    // ---- A fragment: weights row m0+l16, K runs [hf*16..+15]
    union { float4 f[2]; float s[8]; } w0, w1;
    const float* arow = qkv_w + (size_t)(m0 + l16) * NC + kk;
    w0.f[0] = *(const float4*)(arow + hf * 16);
    w0.f[1] = *(const float4*)(arow + hf * 16 + 4);
    w1.f[0] = *(const float4*)(arow + hf * 16 + 8);
    w1.f[1] = *(const float4*)(arow + hf * 16 + 12);
    v16h va;
#pragma unroll
    for (int i = 0; i < 8; ++i) {
      va[i]     = (_Float16)w0.s[i];
      va[i + 8] = (_Float16)w1.s[i];
    }
    // per-lane channel scale/bias (K = hf*16 + i)
    float scv[16], biv[16];
#pragma unroll
    for (int i = 0; i < 16; ++i) {
      int c = kk + hf * 16 + i;
      scv[i] = scb[c]; biv[i] = bib[c];
    }
#pragma unroll
    for (int t = 0; t < 4; ++t) {
      const int n = nb0 + t * 16 + l16;          // B column = pixel
      v16h vb;
#pragma unroll
      for (int i = 0; i < 16; ++i) {
        int c = kk + hf * 16 + i;
        float vv = xb[(size_t)c * NPIX + n];
        vb[i] = (_Float16)(vv * scv[i] + biv[i]);
      }
      acc[t] = WMMA_F16(va, vb, acc[t]);
    }
  }
  // ---- write back (D: lane col = l16, row = r + hf*8)
#pragma unroll
  for (int r = 0; r < 8; ++r) {
    const int m = m0 + r + hf * 8;
    const float qb = qkv_b[m];
#pragma unroll
    for (int t = 0; t < 4; ++t) {
      const int n = nb0 + t * 16 + l16;
      const float v = acc[t][r] + qb;
      if (m0 < 256) {
        qw[((size_t)(b * NPIX + n)) * NC + m] = (_Float16)(v * 0.0625f); // 1/sqrt(256)
      } else if (m0 < 512) {
        kw[((size_t)(b * NPIX + n)) * NC + (m - 256)] = (_Float16)v;
      } else {
        vt[((size_t)(b * NC + (m - 512))) * NPIX + n] = (_Float16)v;
      }
    }
  }
}

// ---------------------------------------------------------------------------
// Kernel 3: flash attention, split-K (flash-decoding). One wave handles
// 16 query rows x 256 channels over KEYS_PER_SPLIT keys; emits normalized
// partial O/l plus (m, l) for the combine pass.
// ---------------------------------------------------------------------------
__global__ __launch_bounds__(128) void attn_kernel(
    const _Float16* __restrict__ qw, const _Float16* __restrict__ kw,
    const _Float16* __restrict__ vt, _Float16* __restrict__ op,
    float* __restrict__ mbuf, float* __restrict__ lbuf) {
  const int lane = threadIdx.x & 31;
  const int w    = threadIdx.x >> 5;
  const int l16  = lane & 15, hf = lane >> 4;
  const int b    = blockIdx.x;
  const int n0   = (blockIdx.y * 4 + w) * 16;
  const int sp   = blockIdx.z;
  const int js0  = sp * KEYS_PER_SPLIT;

  __shared__ __align__(16) float    S[4][16][32];
  __shared__ __align__(16) _Float16 P[4][16][32];
  __shared__ float stat_m[4][16], stat_l[4][16], stat_a[4][16];

  if (lane < 16) { stat_m[w][lane] = -1e30f; stat_l[w][lane] = 0.f; }

  const _Float16* qrow  = qw + ((size_t)(b * NPIX + n0 + l16)) * NC;
  const _Float16* krow0 = kw + (size_t)b * NPIX * NC;
  const _Float16* vtb   = vt + (size_t)b * NC * NPIX;

  v8f acc[16] = {};
  __syncthreads();

  for (int j = js0; j < js0 + KEYS_PER_SPLIT; j += 32) {
    // prefetch next K tile one iteration ahead (global_prefetch_b8)
    __builtin_prefetch((const void*)(krow0 + ((size_t)(j + 32 + lane)) * NC), 0, 1);
    // ---- S = Q(16xC) . K^T(Cx32), two 16x16 D tiles
    v8f s0 = {}, s1 = {};
#pragma unroll
    for (int kc = 0; kc < NC; kc += 32) {
      union { v16h v; v8h h[2]; } qa;
      qa.h[0] = *(const v8h*)(qrow + kc + hf * 16);
      qa.h[1] = *(const v8h*)(qrow + kc + hf * 16 + 8);
      union { v16h v; v8h h[2]; } kb0, kb1;
      const _Float16* kr0 = krow0 + ((size_t)(j + l16)) * NC + kc + hf * 16;
      const _Float16* kr1 = krow0 + ((size_t)(j + 16 + l16)) * NC + kc + hf * 16;
      kb0.h[0] = *(const v8h*)kr0;  kb0.h[1] = *(const v8h*)(kr0 + 8);
      kb1.h[0] = *(const v8h*)kr1;  kb1.h[1] = *(const v8h*)(kr1 + 8);
      s0 = WMMA_F16(qa.v, kb0.v, s0);
      s1 = WMMA_F16(qa.v, kb1.v, s1);
    }
    // ---- stage S in LDS (D layout -> row-major)
#pragma unroll
    for (int r = 0; r < 8; ++r) {
      S[w][r + hf * 8][l16]      = s0[r];
      S[w][r + hf * 8][16 + l16] = s1[r];
    }
    __syncthreads();
    // ---- online softmax: 16 lanes each own one query row
    if (lane < 16) {
      float om = stat_m[w][lane];
      float mx = om;
#pragma unroll
      for (int c = 0; c < 32; ++c) mx = fmaxf(mx, S[w][lane][c]);
      float al = __expf(om - mx);
      float sum = 0.f;
#pragma unroll
      for (int c = 0; c < 32; ++c) {
        float p = __expf(S[w][lane][c] - mx);
        P[w][lane][c] = (_Float16)p;
        sum += p;
      }
      stat_m[w][lane] = mx;
      stat_l[w][lane] = stat_l[w][lane] * al + sum;
      stat_a[w][lane] = al;
    }
    __syncthreads();
    // ---- rescale O accumulators, then O += P . V^T
    float av[8];
#pragma unroll
    for (int r = 0; r < 8; ++r) av[r] = stat_a[w][r + hf * 8];
    union { v16h v; v8h h[2]; } pa;
    pa.h[0] = *(const v8h*)(&P[w][l16][hf * 16]);
    pa.h[1] = *(const v8h*)(&P[w][l16][hf * 16 + 8]);
#pragma unroll
    for (int ct = 0; ct < 16; ++ct) {
#pragma unroll
      for (int r = 0; r < 8; ++r) acc[ct][r] *= av[r];
      union { v16h v; v8h h[2]; } vb;
      const _Float16* vr = vtb + ((size_t)(ct * 16 + l16)) * NPIX + j + hf * 16;
      vb.h[0] = *(const v8h*)vr;  vb.h[1] = *(const v8h*)(vr + 8);
      acc[ct] = WMMA_F16(pa.v, vb.v, acc[ct]);
    }
  }
  // ---- emit partial: op = acc / l  (O(1) magnitude, f16-safe), plus m, l
  if (lane < 16) {
    const int rowg = b * NPIX + n0 + lane;
    mbuf[sp * (NB * NPIX) + rowg] = stat_m[w][lane];
    lbuf[sp * (NB * NPIX) + rowg] = stat_l[w][lane];
    stat_a[w][lane] = 1.0f / stat_l[w][lane];
  }
  __syncthreads();
  float iv[8];
#pragma unroll
  for (int r = 0; r < 8; ++r) iv[r] = stat_a[w][r + hf * 8];
  _Float16* opb = op + (size_t)sp * NB * NPIX * NC;
#pragma unroll
  for (int ct = 0; ct < 16; ++ct) {
#pragma unroll
    for (int r = 0; r < 8; ++r) {
      opb[((size_t)(b * NPIX + n0 + r + hf * 8)) * NC + ct * 16 + l16] =
          (_Float16)(acc[ct][r] * iv[r]);
    }
  }
}

// ---------------------------------------------------------------------------
// Kernel 3b: combine split-K partials with log-sum-exp weights
//   O[row][c] = sum_s a_s * op_s[row][c] / sum_s a_s,  a_s = l_s*exp(m_s - m*)
// ---------------------------------------------------------------------------
__global__ __launch_bounds__(256) void combine_kernel(
    const _Float16* __restrict__ op, const float* __restrict__ mbuf,
    const float* __restrict__ lbuf, _Float16* __restrict__ ob) {
  const int row = blockIdx.x;             // 0 .. NB*NPIX-1
  const int c   = threadIdx.x;            // 0 .. 255
  float ms[SPLIT], ls[SPLIT];
#pragma unroll
  for (int s = 0; s < SPLIT; ++s) {
    ms[s] = mbuf[s * (NB * NPIX) + row];
    ls[s] = lbuf[s * (NB * NPIX) + row];
  }
  float mx = ms[0];
#pragma unroll
  for (int s = 1; s < SPLIT; ++s) mx = fmaxf(mx, ms[s]);
  float a[SPLIT], wsum = 0.f;
#pragma unroll
  for (int s = 0; s < SPLIT; ++s) { a[s] = ls[s] * __expf(ms[s] - mx); wsum += a[s]; }
  const float inv = 1.0f / wsum;
  float o = 0.f;
#pragma unroll
  for (int s = 0; s < SPLIT; ++s) {
    o += a[s] * (float)op[((size_t)s * NB * NPIX + row) * NC + c];
  }
  ob[(size_t)row * NC + c] = (_Float16)(o * inv);
}

// ---------------------------------------------------------------------------
// Kernel 4: out = x + proj_w . o + proj_b   (WMMA GEMM + residual)
// ---------------------------------------------------------------------------
__global__ __launch_bounds__(128) void proj_kernel(
    const float* __restrict__ x, const float* __restrict__ proj_w,
    const float* __restrict__ proj_b, const _Float16* __restrict__ ob,
    float* __restrict__ out) {
  const int lane = threadIdx.x & 31;
  const int wid  = threadIdx.x >> 5;
  const int l16  = lane & 15, hf = lane >> 4;
  const int m0   = blockIdx.x * 64 + wid * 16;
  const int ng0  = blockIdx.y * 64;
  const int b    = ng0 >> 12;
  const int nb0  = ng0 & (NPIX - 1);
  const _Float16* obb = ob + (size_t)b * NPIX * NC;

  v8f acc[4] = {};
  for (int kk = 0; kk < NC; kk += 32) {
    union { float4 f[2]; float s[8]; } w0, w1;
    const float* arow = proj_w + (size_t)(m0 + l16) * NC + kk;
    w0.f[0] = *(const float4*)(arow + hf * 16);
    w0.f[1] = *(const float4*)(arow + hf * 16 + 4);
    w1.f[0] = *(const float4*)(arow + hf * 16 + 8);
    w1.f[1] = *(const float4*)(arow + hf * 16 + 12);
    v16h va;
#pragma unroll
    for (int i = 0; i < 8; ++i) {
      va[i]     = (_Float16)w0.s[i];
      va[i + 8] = (_Float16)w1.s[i];
    }
#pragma unroll
    for (int t = 0; t < 4; ++t) {
      union { v16h v; v8h h[2]; } u;
      const _Float16* br = obb + ((size_t)(nb0 + t * 16 + l16)) * NC + kk + hf * 16;
      u.h[0] = *(const v8h*)br;  u.h[1] = *(const v8h*)(br + 8);
      acc[t] = WMMA_F16(va, u.v, acc[t]);
    }
  }
#pragma unroll
  for (int r = 0; r < 8; ++r) {
    const int m = m0 + r + hf * 8;
    const float pb = proj_b[m];
#pragma unroll
    for (int t = 0; t < 4; ++t) {
      const int n = nb0 + t * 16 + l16;
      const size_t idx = ((size_t)(b * NC + m)) * NPIX + n;
      out[idx] = x[idx] + acc[t][r] + pb;
    }
  }
}

// ---------------------------------------------------------------------------
extern "C" void kernel_launch(void* const* d_in, const int* in_sizes, int n_in,
                              void* d_out, int out_size, void* d_ws, size_t ws_size,
                              hipStream_t stream) {
  (void)in_sizes; (void)n_in; (void)out_size; (void)ws_size;
  const float* x      = (const float*)d_in[0];
  const float* gn_w   = (const float*)d_in[1];
  const float* gn_b   = (const float*)d_in[2];
  const float* qkv_w  = (const float*)d_in[3];
  const float* qkv_b  = (const float*)d_in[4];
  const float* proj_w = (const float*)d_in[5];
  const float* proj_b = (const float*)d_in[6];
  float* out = (float*)d_out;

  const size_t NH = (size_t)NB * NPIX * NC;     // elements per f16 tensor
  char* ws = (char*)d_ws;
  float* scale = (float*)ws;                    // NB*NC floats
  float* bias  = scale + NB * NC;               // NB*NC floats
  _Float16* qw = (_Float16*)(ws + 8192);
  _Float16* kw = qw + NH;
  _Float16* vt = kw + NH;
  _Float16* op = vt + NH;                       // SPLIT partials
  _Float16* ob = op + (size_t)SPLIT * NH;
  float* mbuf  = (float*)(ob + NH);             // SPLIT*NB*NPIX floats
  float* lbuf  = mbuf + (size_t)SPLIT * NB * NPIX;

  gn_stats_kernel<<<dim3(NB * NG), 256, 0, stream>>>(x, gn_w, gn_b, scale, bias);
  qkv_kernel<<<dim3(12, 256), 128, 0, stream>>>(x, qkv_w, qkv_b, scale, bias, qw, kw, vt);
  attn_kernel<<<dim3(NB, 64, SPLIT), 128, 0, stream>>>(qw, kw, vt, op, mbuf, lbuf);
  combine_kernel<<<dim3(NB * NPIX), 256, 0, stream>>>(op, mbuf, lbuf, ob);
  proj_kernel<<<dim3(4, 256), 128, 0, stream>>>(x, proj_w, proj_b, ob, out);
}
